// FourierBlock_7438883356760
// MI455X (gfx1250) — matrix operand order repaired and checked by reading, not compiled, not run on hardware
//
#include <hip/hip_runtime.h>
#include <hip/hip_bf16.h>

// ---------------------------------------------------------------------------
// FNO block for MI455X (gfx1250, wave32, WMMA + async-to-LDS).
// Problem: B=16, L=4096, H=8, E=64, C=H*E=512, MODES=64, O=OUT_CH/8=64.
// Pipeline: [init bases] -> [fwd partial DFT gemm] -> [mode mix] -> [inv DFT gemm]
// Memory-bound (~300 MB @ 23.3 TB/s); compute (17.5 GFLOP) via f16 WMMA, f32 acc.
// ---------------------------------------------------------------------------

typedef __attribute__((ext_vector_type(16))) _Float16 v16h;
typedef __attribute__((ext_vector_type(8)))  _Float16 v8h;
typedef __attribute__((ext_vector_type(8)))  float    v8f;
typedef __attribute__((ext_vector_type(4)))  int      v4i;

// pointer-to-AS1/AS3 v4i (attribute applies to pointee)
typedef __attribute__((address_space(1))) v4i* as1_v4i_p;
typedef __attribute__((address_space(3))) v4i* as3_v4i_p;

#define NB   16
#define NL   4096
#define NH   8
#define NE   64
#define NC   512      // NH*NE
#define NM   64       // modes
#define NM2  128      // interleaved re/im coefficients
#define NO   64       // OUT_CH/8
#define NR   8192     // NB*NH*NO  == NB*NC output rows

// Workspace layout (bytes); total 8 MB.
#define WS_WFT  (0u)          // f16 [128][4096]   forward basis, m2-major (K contiguous)
#define WS_BIL  (1u << 20)    // f16 [4096][128]   inverse basis, l-major (K contiguous)
#define WS_SEL  (2u << 20)    // f32 [16][512][128] selected modes (re/im interleaved)
#define WS_COEF (6u << 20)    // f16 [8192][128]   mixed-mode coefficients

#if __has_builtin(__builtin_amdgcn_global_load_async_to_lds_b128) && \
    __has_builtin(__builtin_amdgcn_s_wait_asynccnt)
#define FB_HAS_ASYNC_LDS 1
#else
#define FB_HAS_ASYNC_LDS 0
#endif

// 16B global -> LDS copy; async (ASYNCcnt-tracked) when available.
__device__ __forceinline__ void fb_copy_b128(const _Float16* g, _Float16* l) {
#if FB_HAS_ASYNC_LDS
  __builtin_amdgcn_global_load_async_to_lds_b128((as1_v4i_p)g, (as3_v4i_p)l, 0, 0);
#else
  *(v8h*)l = *(const v8h*)g;
#endif
}

__device__ __forceinline__ void fb_wait_async() {
#if FB_HAS_ASYNC_LDS
  __builtin_amdgcn_s_wait_asynccnt(0);
#endif
}

__device__ __forceinline__ v8f v8f_zero() {
  v8f z;
  #pragma unroll
  for (int e = 0; e < 8; ++e) z[e] = 0.0f;
  return z;
}

// A-fragment (16x32 f16): lanes 0-15 rows M, halves j -> K=hi*16+j.
// Two contiguous 16B LDS reads.
__device__ __forceinline__ v16h load_a_frag(const _Float16* row, int hi) {
  union { v16h v; v8h h[2]; } u;
  u.h[0] = *(const v8h*)(row + hi * 16);
  u.h[1] = *(const v8h*)(row + hi * 16 + 8);
  return u.v;
}

// ---------------------------------------------------------------------------
// Kernel 0: build f16 DFT bases in workspace.
//   WfT[m2][l]: m2=2m -> cos(2pi m l/L), m2=2m+1 -> -sin(...)   (rfft)
//   BiL[l][k2]: k2=2m -> s_m*cos, k2=2m+1 -> -s_m*sin, s_m=(m?2:1)/L (irfft)
// Phase computed as exact integer (m*l mod 4096) to keep sincos args small.
// ---------------------------------------------------------------------------
__global__ void __launch_bounds__(256)
fb_init_basis(_Float16* __restrict__ wft, _Float16* __restrict__ bil) {
  int idx = blockIdx.x * 256 + threadIdx.x;            // 0 .. 2*524288-1
  const float TWO_PI = 6.283185307179586f;
  if (idx < 128 * 4096) {
    int m2 = idx >> 12, l = idx & 4095, m = m2 >> 1;
    float th = TWO_PI * (float)((m * l) & 4095) * (1.0f / 4096.0f);
    float s, c;
    __sincosf(th, &s, &c);
    wft[m2 * 4096 + l] = (_Float16)((m2 & 1) ? -s : c);
  } else {
    idx -= 128 * 4096;
    int l = idx >> 7, k2 = idx & 127, m = k2 >> 1;
    float th = TWO_PI * (float)((m * l) & 4095) * (1.0f / 4096.0f);
    float s, c;
    __sincosf(th, &s, &c);
    float sc = (m == 0 ? 1.0f : 2.0f) * (1.0f / 4096.0f);
    bil[l * 128 + k2] = (_Float16)(((k2 & 1) ? -s : c) * sc);
  }
}

// ---------------------------------------------------------------------------
// Kernel 1: forward partial DFT.  sel[b,c,m2] = sum_l q[b,l,c] * WfT[m2][l]
// Workgroup = (b, 64-channel block). 8 waves = 4 c-tiles x 2 m-halves.
// LDS tiles stored K(=l)-contiguous so WMMA fragments are vector LDS reads.
// ---------------------------------------------------------------------------
__global__ void __launch_bounds__(256)
fb_fwd_dft(const float* __restrict__ q, const _Float16* __restrict__ wft,
           float* __restrict__ sel) {
  __shared__ _Float16 sQ[64 * 32];   // [c_local][l_local]
  __shared__ _Float16 sB[128 * 32];  // [m2][l_local]

  const int b    = blockIdx.x;
  const int cblk = blockIdx.y;            // 0..7 -> 64 channels each
  const int t    = threadIdx.x;
  const int lane = t & 31, wv = t >> 5;
  const int hi = lane >> 4, n = lane & 15;
  const int cw = wv & 3;                  // c-tile within the 64-channel block
  const int mh = wv >> 2;                 // m-half: tiles mh*4 .. mh*4+3

  v8f acc[4];
  #pragma unroll
  for (int i = 0; i < 4; ++i) acc[i] = v8f_zero();

  const int cq = t & 63, i0 = (t >> 6) * 8;    // staging coords for q tile
  const int m2s = t >> 1, chs = (t & 1) * 16;  // staging coords for basis tile

  for (int l0 = 0; l0 < NL; l0 += 32) {
    __syncthreads();
    // Stage forward-basis tile (straight copy, K-contiguous): async -> LDS.
    fb_copy_b128(&wft[m2s * 4096 + l0 + chs],     &sB[m2s * 32 + chs]);
    fb_copy_b128(&wft[m2s * 4096 + l0 + chs + 8], &sB[m2s * 32 + chs + 8]);
    // Stage q tile (f32 -> f16 conversion), transposing to [c][l].
    {
      const float* gq = q + ((size_t)b * NL + l0) * NC + cblk * 64 + cq;
      if (l0 + 32 < NL) __builtin_prefetch(gq + 32 * NC, 0, 0);  // global_prefetch_b8
      #pragma unroll
      for (int i = 0; i < 8; ++i)
        sQ[cq * 32 + i0 + i] = (_Float16)gq[(size_t)(i0 + i) * NC];
    }
    fb_wait_async();
    __syncthreads();

    const _Float16* arow = &sQ[(cw * 16 + n) * 32];
    v16h af = load_a_frag(arow, hi);
    #pragma unroll
    for (int mt = 0; mt < 4; ++mt) {
      // B 32x16: lane n -> column m2, halves j -> K = hi*16 + j (contiguous 32B)
      v16h bf = *(const v16h*)&sB[((mh * 4 + mt) * 16 + n) * 32 + hi * 16];
      acc[mt] = __builtin_amdgcn_wmma_f32_16x16x32_f16(
          false, af, false, bf, (short)0, acc[mt], false, false);
    }
  }

  // C/D layout: VGPR v -> row v + 8*hi, lane&15 -> column.
  #pragma unroll
  for (int mt = 0; mt < 4; ++mt) {
    const int m2 = (mh * 4 + mt) * 16 + n;
    const int c0 = cblk * 64 + cw * 16 + hi * 8;
    float* dst = sel + ((size_t)b * NC + c0) * NM2 + m2;
    #pragma unroll
    for (int v = 0; v < 8; ++v) dst[(size_t)v * NM2] = acc[mt][v];
  }
}

// ---------------------------------------------------------------------------
// Kernel 2: complex mode mixing (tiny: 0.27 GFLOP -> VALU).
// out[b,h,o,m] = sum_i sel[b,h,i,m] * (w1+i*w2)[h,i,o,m]; emit f16 coef rows.
// ---------------------------------------------------------------------------
__global__ void __launch_bounds__(256)
fb_mix_modes(const float* __restrict__ sel, const float* __restrict__ w1,
             const float* __restrict__ w2, _Float16* __restrict__ coef) {
  const int t = blockIdx.x * 256 + threadIdx.x;   // 0 .. 524287
  const int m = t & 63;
  const int o = (t >> 6) & 63;
  const int h = (t >> 12) & 7;
  const int b = t >> 15;

  const float* sb  = sel + ((size_t)b * NC + h * 64) * NM2 + 2 * m; // i-stride 128
  const float* w1p = w1 + (size_t)h * 262144 + (size_t)o * 64 + m;  // i-stride 4096
  const float* w2p = w2 + (size_t)h * 262144 + (size_t)o * 64 + m;

  float re = 0.0f, im = 0.0f;
  #pragma unroll 4
  for (int i = 0; i < 64; ++i) {
    float sr = sb[i * NM2], si = sb[i * NM2 + 1];
    float a = w1p[(size_t)i * 4096], c = w2p[(size_t)i * 4096];
    re = fmaf(sr, a, re); re = fmaf(-si, c, re);
    im = fmaf(sr, c, im); im = fmaf(si, a, im);
  }
  _Float16* cp = coef + (((size_t)(b * NH + h)) * NO + o) * NM2 + 2 * m;
  cp[0] = (_Float16)re;
  cp[1] = (_Float16)im;
}

// ---------------------------------------------------------------------------
// Kernel 3: inverse partial DFT.  out[r,l] = sum_k coef[r][k] * BiL[l][k]
// 128x128 output tile per workgroup, full K=128 staged once in LDS (64 KB)
// via async-to-LDS.  8 waves: wave w owns rows w*16..15, eight 16-wide l-tiles.
// ---------------------------------------------------------------------------
__global__ void __launch_bounds__(256)
fb_inv_dft(const _Float16* __restrict__ coef, const _Float16* __restrict__ bil,
           float* __restrict__ out) {
  __shared__ _Float16 As[128 * 128];  // coef rows   [r_local][k]
  __shared__ _Float16 Bs[128 * 128];  // basis cols  [l_local][k]

  const int r0 = blockIdx.x * 128;    // 64 row blocks
  const int l0 = blockIdx.y * 128;    // 32 col blocks
  const int t  = threadIdx.x;
  const int lane = t & 31, wv = t >> 5;
  const int hi = lane >> 4, n = lane & 15;

  {
    const int r = t >> 1, ch = (t & 1) * 64;
    const _Float16* gA = coef + (size_t)(r0 + r) * NM2 + ch;
    const _Float16* gB = bil  + (size_t)(l0 + r) * NM2 + ch;
    #pragma unroll
    for (int u = 0; u < 8; ++u) {
      fb_copy_b128(gA + u * 8, &As[r * 128 + ch + u * 8]);
      fb_copy_b128(gB + u * 8, &Bs[r * 128 + ch + u * 8]);
    }
    fb_wait_async();
  }
  __syncthreads();

  v8f acc[8];
  #pragma unroll
  for (int i = 0; i < 8; ++i) acc[i] = v8f_zero();

  #pragma unroll
  for (int k0 = 0; k0 < 128; k0 += 32) {
    v16h af = load_a_frag(&As[(wv * 16 + n) * 128 + k0], hi);
    #pragma unroll
    for (int lt = 0; lt < 8; ++lt) {
      v16h bf = *(const v16h*)&Bs[(lt * 16 + n) * 128 + k0 + hi * 16];
      acc[lt] = __builtin_amdgcn_wmma_f32_16x16x32_f16(
          false, af, false, bf, (short)0, acc[lt], false, false);
    }
  }

  #pragma unroll
  for (int lt = 0; lt < 8; ++lt) {
    const int col = l0 + lt * 16 + n;
    float* dst = out + (size_t)(r0 + wv * 16 + hi * 8) * NL + col;
    #pragma unroll
    for (int v = 0; v < 8; ++v) dst[(size_t)v * NL] = acc[lt][v];
  }
}

// ---------------------------------------------------------------------------
extern "C" void kernel_launch(void* const* d_in, const int* in_sizes, int n_in,
                              void* d_out, int out_size, void* d_ws, size_t ws_size,
                              hipStream_t stream) {
  (void)in_sizes; (void)n_in; (void)out_size; (void)ws_size;
  const float* q  = (const float*)d_in[0];  // (16,4096,8,64)  -- k,v,mask unused
  const float* w1 = (const float*)d_in[4];  // (8,64,64,64)
  const float* w2 = (const float*)d_in[5];
  float* out = (float*)d_out;               // (16,8,64,4096)

  _Float16* wft  = (_Float16*)((char*)d_ws + WS_WFT);
  _Float16* bil  = (_Float16*)((char*)d_ws + WS_BIL);
  float*    sel  = (float*)   ((char*)d_ws + WS_SEL);
  _Float16* coef = (_Float16*)((char*)d_ws + WS_COEF);

  fb_init_basis<<<4096, 256, 0, stream>>>(wft, bil);
  fb_fwd_dft  <<<dim3(NB, 8), 256, 0, stream>>>(q, wft, sel);
  fb_mix_modes<<<2048, 256, 0, stream>>>(sel, w1, w2, coef);
  fb_inv_dft  <<<dim3(NR / 128, NL / 128), 256, 0, stream>>>(coef, bil, out);
}
